// MyModel_15728170238623
// MI455X (gfx1250) — matrix operand order repaired and checked by reading, not compiled, not run on hardware
//
#include <hip/hip_runtime.h>
#include <math.h>

typedef __attribute__((ext_vector_type(16))) _Float16 v16h;
typedef __attribute__((ext_vector_type(8)))  _Float16 v8h;
typedef __attribute__((ext_vector_type(8)))  float    v8f;

#define NEG_SLOPE 0.2f
#define EPSV 1e-5f

__device__ __forceinline__ float gelu_exact(float x) {
    return 0.5f * x * (1.0f + erff(x * 0.7071067811865476f));
}
__device__ __forceinline__ float bn_eval(float x, float g, float b, float m, float v) {
    return (x - m) * rsqrtf(v + EPSV) * g + b;
}
// float atomic max via monotonic int/uint ordering trick
__device__ __forceinline__ void atomicMaxF32(float* addr, float val) {
    if (val >= 0.0f) atomicMax((int*)addr, __float_as_int(val));
    else             atomicMin((unsigned int*)addr, __float_as_uint(val));
}

// ---------------------------------------------------------------- utilities
__global__ void fill_kernel(float* __restrict__ p, float v, size_t n) {
    size_t i = (size_t)blockIdx.x * blockDim.x + threadIdx.x;
    if (i < n) p[i] = v;
}

// f32 [M x K] -> f16 [M x Kpad], zero padded K tail
__global__ void cvt_pad_f16(const float* __restrict__ src, _Float16* __restrict__ dst,
                            int M, int K, int Kpad) {
    size_t i = (size_t)blockIdx.x * blockDim.x + threadIdx.x;
    if (i >= (size_t)M * Kpad) return;
    int m = (int)(i / Kpad);
    int k = (int)(i - (size_t)m * Kpad);
    dst[i] = (k < K) ? (_Float16)src[(size_t)m * K + k] : (_Float16)0.0f;
}
// f32 W [K x N] -> f16 Wt [N x Kpad] (transposed, zero padded K tail)
__global__ void cvt_wt_f16(const float* __restrict__ W, _Float16* __restrict__ Wt,
                           int K, int Ncols, int Kpad) {
    size_t i = (size_t)blockIdx.x * blockDim.x + threadIdx.x;
    if (i >= (size_t)Ncols * Kpad) return;
    int n = (int)(i / Kpad);
    int k = (int)(i - (size_t)n * Kpad);
    Wt[i] = (k < K) ? (_Float16)W[(size_t)k * Ncols + n] : (_Float16)0.0f;
}

// ------------------------------------------------- self-loop attr (mean of incoming)
__global__ void loop_attr_acc(const int* __restrict__ etgt, const float* __restrict__ eattr,
                              float* __restrict__ lsum, float* __restrict__ cnt, int E) {
    int e = blockIdx.x * blockDim.x + threadIdx.x;
    if (e >= E) return;
    int t = etgt[e];
    atomicAdd(&cnt[t], 1.0f);
#pragma unroll
    for (int k = 0; k < 12; ++k)
        atomicAdd(&lsum[(size_t)t * 12 + k], eattr[(size_t)e * 12 + k]);
}
__global__ void loop_attr_fin(float* __restrict__ lsum, const float* __restrict__ cnt, int N) {
    size_t i = (size_t)blockIdx.x * blockDim.x + threadIdx.x;
    if (i >= (size_t)N * 12) return;
    lsum[i] /= fmaxf(cnt[i / 12], 1.0f);
}

// ------------------------------------------------- WMMA GEMM on pre-converted operands
// D = A(f16 M x Kpad) * Wt(f16 N x Kpad, transposed) + bias.  One wave32 per 16x16 tile.
// A fragment per ISA 16-bit A 16x32 layout: lane group K-shift 8; element K = (e&7)+(e>=8?16:0).
// B fragment: same layout as A (lane group K-shift 8).  All loads 16B aligned b128.
__global__ void gemm_wmma_f16(const _Float16* __restrict__ A, const _Float16* __restrict__ Wt,
                              const float* __restrict__ bias, float* __restrict__ D,
                              int M, int Kpad, int Ncols) {
    const int lane   = threadIdx.x & 31;
    const int wave   = (blockIdx.x * blockDim.x + threadIdx.x) >> 5;
    const int tilesN = Ncols >> 4;
    const int nTiles = (M >> 4) * tilesN;
    if (wave >= nTiles) return;
    const int tileM = wave / tilesN;
    const int tileN = wave - tileM * tilesN;
    const int rowA  = tileM * 16 + (lane & 15);
    const int colB  = tileN * 16 + (lane & 15);
    const _Float16* arow = A  + (size_t)rowA * Kpad + ((lane & 16) ? 8  : 0);
    const _Float16* brow = Wt + (size_t)colB * Kpad + ((lane & 16) ? 8 : 0);
    v8f acc = {};
    for (int k0 = 0; k0 < Kpad; k0 += 32) {
        __builtin_prefetch(arow + k0 + 256, 0, 0);
        union { v16h v; v8h h[2]; } a, b;
        a.h[0] = *(const v8h*)(arow + k0);        // K = kA + 0..7
        a.h[1] = *(const v8h*)(arow + k0 + 16);   // K = kA + 16..23
        b.h[0] = *(const v8h*)(brow + k0);        // K = kB + 0..7
        b.h[1] = *(const v8h*)(brow + k0 + 16);   // K = kB + 16..23
        acc = __builtin_amdgcn_wmma_f32_16x16x32_f16(false, a.v, false, b.v,
                                                     (short)0, acc, false, false);
    }
    const int rBase = tileM * 16 + ((lane & 16) ? 8 : 0);
#pragma unroll
    for (int r = 0; r < 8; ++r)
        D[(size_t)(rBase + r) * Ncols + colB] = acc[r] + bias[colB];
}

// Same GEMM with fused bias + BN + exact GELU on store (fingerprint branch).
__global__ void gemm_wmma_f16_bngelu(const _Float16* __restrict__ A, const _Float16* __restrict__ Wt,
                                     const float* __restrict__ bias,
                                     const float* __restrict__ bg, const float* __restrict__ bb,
                                     const float* __restrict__ bm, const float* __restrict__ bv,
                                     float* __restrict__ D, int M, int Kpad, int Ncols) {
    const int lane   = threadIdx.x & 31;
    const int wave   = (blockIdx.x * blockDim.x + threadIdx.x) >> 5;
    const int tilesN = Ncols >> 4;
    const int nTiles = (M >> 4) * tilesN;
    if (wave >= nTiles) return;
    const int tileM = wave / tilesN;
    const int tileN = wave - tileM * tilesN;
    const int rowA  = tileM * 16 + (lane & 15);
    const int colB  = tileN * 16 + (lane & 15);
    const _Float16* arow = A  + (size_t)rowA * Kpad + ((lane & 16) ? 8  : 0);
    const _Float16* brow = Wt + (size_t)colB * Kpad + ((lane & 16) ? 8 : 0);
    v8f acc = {};
    for (int k0 = 0; k0 < Kpad; k0 += 32) {
        __builtin_prefetch(arow + k0 + 256, 0, 0);
        union { v16h v; v8h h[2]; } a, b;
        a.h[0] = *(const v8h*)(arow + k0);
        a.h[1] = *(const v8h*)(arow + k0 + 16);
        b.h[0] = *(const v8h*)(brow + k0);
        b.h[1] = *(const v8h*)(brow + k0 + 16);
        acc = __builtin_amdgcn_wmma_f32_16x16x32_f16(false, a.v, false, b.v,
                                                     (short)0, acc, false, false);
    }
    const int rBase = tileM * 16 + ((lane & 16) ? 8 : 0);
    const float g = bg[colB], b2 = bb[colB], m2 = bm[colB], v2 = bv[colB], bi = bias[colB];
#pragma unroll
    for (int r = 0; r < 8; ++r) {
        float v = acc[r] + bi;
        v = gelu_exact(bn_eval(v, g, b2, m2, v2));
        D[(size_t)(rBase + r) * Ncols + colB] = v;
    }
}

// ------------------------------------------------- edge pass 1: logits + segment max
// One wave32 per edge. lane handles channels [4*lane, 4*lane+4) -> head = lane>>3.
__global__ void gat_edge_logit(const int* __restrict__ esrc, const int* __restrict__ etgt,
                               const float* __restrict__ eattr, const float* __restrict__ loop_attr,
                               const float* __restrict__ xl, const float* __restrict__ xr,
                               const float* __restrict__ We, const float* __restrict__ att,
                               float* __restrict__ logit, float* __restrict__ mx,
                               int E, int N) {
    __shared__ float sWe[12 * 128];
    __shared__ float sAtt[128];
    for (int i = threadIdx.x; i < 12 * 128; i += blockDim.x) sWe[i] = We[i];
    for (int i = threadIdx.x; i < 128;      i += blockDim.x) sAtt[i] = att[i];
    __syncthreads();
    const int lane = threadIdx.x & 31;
    const int e    = (blockIdx.x * blockDim.x + threadIdx.x) >> 5;
    if (e >= E + N) return;
    int s, t; const float* ea;
    if (e < E) { s = esrc[e]; t = etgt[e]; ea = eattr + (size_t)e * 12; }
    else       { s = t = e - E;            ea = loop_attr + (size_t)(e - E) * 12; }
    float eav = (lane < 12) ? ea[lane] : 0.0f;
    float part = 0.0f;
#pragma unroll
    for (int j = 0; j < 4; ++j) {
        int c = lane * 4 + j;
        float em = 0.0f;
#pragma unroll
        for (int k = 0; k < 12; ++k)
            em = fmaf(__shfl(eav, k, 32), sWe[k * 128 + c], em);
        float m = xl[(size_t)s * 128 + c] + xr[(size_t)t * 128 + c] + em;
        m = (m > 0.0f) ? m : NEG_SLOPE * m;
        part = fmaf(m, sAtt[c], part);
    }
    part += __shfl_xor(part, 1, 32);
    part += __shfl_xor(part, 2, 32);
    part += __shfl_xor(part, 4, 32);
    if ((lane & 7) == 0) {
        int h = lane >> 3;
        logit[(size_t)e * 4 + h] = part;
        atomicMaxF32(&mx[(size_t)t * 4 + h], part);
    }
}

// ------------------------------------------------- edge pass 2: exp + segment sum
__global__ void gat_edge_exp(const int* __restrict__ etgt, float* __restrict__ logit,
                             const float* __restrict__ mx, float* __restrict__ denom,
                             int E, int N) {
    int e = blockIdx.x * blockDim.x + threadIdx.x;
    if (e >= E + N) return;
    int t = (e < E) ? etgt[e] : (e - E);
#pragma unroll
    for (int h = 0; h < 4; ++h) {
        float ex = __expf(logit[(size_t)e * 4 + h] - mx[(size_t)t * 4 + h]);
        logit[(size_t)e * 4 + h] = ex;
        atomicAdd(&denom[(size_t)t * 4 + h], ex);
    }
}

// ------------------------------------------------- edge pass 3: alpha-weighted aggregation
__global__ void gat_edge_aggr(const int* __restrict__ esrc, const int* __restrict__ etgt,
                              const float* __restrict__ exw, const float* __restrict__ denom,
                              const float* __restrict__ xl, float* __restrict__ outAcc,
                              int E, int N) {
    const int lane = threadIdx.x & 31;
    const int e    = (blockIdx.x * blockDim.x + threadIdx.x) >> 5;
    if (e >= E + N) return;
    int s, t;
    if (e < E) { s = esrc[e]; t = etgt[e]; } else { s = t = e - E; }
    int h = lane >> 3;
    float alpha = exw[(size_t)e * 4 + h] / denom[(size_t)t * 4 + h];
#pragma unroll
    for (int j = 0; j < 4; ++j) {
        int c = lane * 4 + j;
        atomicAdd(&outAcc[(size_t)t * 128 + c], alpha * xl[(size_t)s * 128 + c]);
    }
}

// ------------------------------------------------- node post: bias + BN + GELU + mean-pool scatter
__global__ void gat_node_post(const float* __restrict__ outAcc, const float* __restrict__ gbias,
                              const float* __restrict__ bg, const float* __restrict__ bb,
                              const float* __restrict__ bm, const float* __restrict__ bv,
                              const int* __restrict__ batch, float* __restrict__ pool,
                              float* __restrict__ gcnt, int N) {
    size_t i = (size_t)blockIdx.x * blockDim.x + threadIdx.x;
    if (i >= (size_t)N * 128) return;
    int n = (int)(i >> 7);
    int c = (int)(i & 127);
    float v = outAcc[i] + gbias[c];
    v = gelu_exact(bn_eval(v, bg[c], bb[c], bm[c], bv[c]));
    int g = batch[n];
    atomicAdd(&pool[(size_t)g * 128 + c], v);
    if (c == 0) atomicAdd(&gcnt[g], 1.0f);
}

// ------------------------------------------------- fusion head: one wave32 per graph
__global__ void fuse_head(const float* __restrict__ pool, const float* __restrict__ gcnt,
                          const float* __restrict__ fpact, const float* __restrict__ cx,
                          const float* __restrict__ f1w, const float* __restrict__ f1b,
                          const float* __restrict__ fg, const float* __restrict__ fb,
                          const float* __restrict__ fm, const float* __restrict__ fv,
                          const float* __restrict__ f2w, const float* __restrict__ f2b,
                          float* __restrict__ out) {
    __shared__ float z[202];
    int b = blockIdx.x, t = threadIdx.x;
    float inv = 1.0f / fmaxf(gcnt[b], 1.0f);
    for (int k = t; k < 202; k += 32) {
        float v;
        if (k < 128)      v = pool[(size_t)b * 128 + k] * inv;
        else if (k < 192) v = fpact[(size_t)b * 64 + (k - 128)];
        else              v = cx[(size_t)b * 10 + (k - 192)];
        z[k] = v;
    }
    __syncthreads();
    float s = f1b[t];
    for (int k = 0; k < 202; ++k) s = fmaf(z[k], f1w[k * 32 + t], s);
    s = gelu_exact(bn_eval(s, fg[t], fb[t], fm[t], fv[t]));
    float r = s * f2w[t];
    r += __shfl_xor(r, 1, 32);
    r += __shfl_xor(r, 2, 32);
    r += __shfl_xor(r, 4, 32);
    r += __shfl_xor(r, 8, 32);
    r += __shfl_xor(r, 16, 32);
    if (t == 0) out[b] = r + f2b[0];
}

// =================================================================== launch
static inline int cdiv_l(long a, long b) { return (int)((a + b - 1) / b); }

extern "C" void kernel_launch(void* const* d_in, const int* in_sizes, int n_in,
                              void* d_out, int out_size, void* d_ws, size_t ws_size,
                              hipStream_t stream) {
    const float* x     = (const float*)d_in[0];
    const int*   eidx  = (const int*)  d_in[1];
    const float* eattr = (const float*)d_in[2];
    const int*   batch = (const int*)  d_in[3];
    const float* fpx   = (const float*)d_in[4];
    const float* cx    = (const float*)d_in[5];
    const float* glw   = (const float*)d_in[6];
    const float* glb   = (const float*)d_in[7];
    const float* grw   = (const float*)d_in[8];
    const float* grb   = (const float*)d_in[9];
    const float* gew   = (const float*)d_in[10];
    const float* gatt  = (const float*)d_in[11];
    const float* gbias = (const float*)d_in[12];
    const float* bn4g  = (const float*)d_in[13];
    const float* bn4b  = (const float*)d_in[14];
    const float* bn4m  = (const float*)d_in[15];
    const float* bn4v  = (const float*)d_in[16];
    const float* fpw   = (const float*)d_in[17];
    const float* fpb   = (const float*)d_in[18];
    const float* fpbng = (const float*)d_in[19];
    const float* fpbnb = (const float*)d_in[20];
    const float* fpbnm = (const float*)d_in[21];
    const float* fpbnv = (const float*)d_in[22];
    const float* f1w   = (const float*)d_in[23];
    const float* f1b   = (const float*)d_in[24];
    const float* fbng  = (const float*)d_in[25];
    const float* fbnb  = (const float*)d_in[26];
    const float* fbnm  = (const float*)d_in[27];
    const float* fbnv  = (const float*)d_in[28];
    const float* f2w   = (const float*)d_in[29];
    const float* f2b   = (const float*)d_in[30];

    const int N = in_sizes[3];          // nodes
    const int E = in_sizes[1] / 2;      // edges
    const int B = in_sizes[5] / 10;     // graphs
    const int Etot = E + N;
    const int* esrc = eidx;
    const int* etgt = eidx + E;
    const int FPK    = in_sizes[17] / 64;            // 3387
    const int FPKpad = ((FPK + 31) / 32) * 32;       // 3392
    const int KX     = 54, KXpad = 64;

    // workspace layout (f32 region, then f16 region; all region sizes multiple of 16B)
    float* ws      = (float*)d_ws;
    float* xl      = ws;                                 // N*128
    float* xr      = xl      + (size_t)N * 128;          // N*128
    float* outAcc  = xr      + (size_t)N * 128;          // N*128
    float* lsum    = outAcc  + (size_t)N * 128;          // N*12 (-> loop_attr)
    float* cnt     = lsum    + (size_t)N * 12;           // N
    float* mx      = cnt     + (size_t)N;                // N*4
    float* denom   = mx      + (size_t)N * 4;            // N*4
    float* elog    = denom   + (size_t)N * 4;            // (E+N)*4 (logit -> exp)
    float* pool    = elog    + (size_t)Etot * 4;         // B*128
    float* gcnt    = pool    + (size_t)B * 128;          // B
    float* fpact   = gcnt    + (size_t)B;                // B*64
    _Float16* xpad = (_Float16*)(fpact + (size_t)B * 64);// N*KXpad
    _Float16* wlT  = xpad + (size_t)N * KXpad;           // 128*KXpad
    _Float16* wrT  = wlT  + (size_t)128 * KXpad;         // 128*KXpad
    _Float16* fpxp = wrT  + (size_t)128 * KXpad;         // B*FPKpad
    _Float16* fpwT = fpxp + (size_t)B * FPKpad;          // 64*FPKpad

    // --- init accumulators (every call: deterministic, graph-capture safe)
    fill_kernel<<<cdiv_l((long)N * 128, 256), 256, 0, stream>>>(outAcc, 0.0f, (size_t)N * 128);
    fill_kernel<<<cdiv_l((long)N * 12, 256),  256, 0, stream>>>(lsum,   0.0f, (size_t)N * 12);
    fill_kernel<<<cdiv_l((long)N, 256),       256, 0, stream>>>(cnt,    0.0f, (size_t)N);
    fill_kernel<<<cdiv_l((long)N * 4, 256),   256, 0, stream>>>(mx, -INFINITY, (size_t)N * 4);
    fill_kernel<<<cdiv_l((long)N * 4, 256),   256, 0, stream>>>(denom,  0.0f, (size_t)N * 4);
    fill_kernel<<<cdiv_l((long)B * 128, 256), 256, 0, stream>>>(pool,   0.0f, (size_t)B * 128);
    fill_kernel<<<cdiv_l((long)B, 256),       256, 0, stream>>>(gcnt,   0.0f, (size_t)B);

    // --- convert/pad/transpose GEMM operands to f16 (streaming, coalesced)
    cvt_pad_f16<<<cdiv_l((long)N * KXpad, 256), 256, 0, stream>>>(x, xpad, N, KX, KXpad);
    cvt_wt_f16 <<<cdiv_l((long)128 * KXpad, 256), 256, 0, stream>>>(glw, wlT, KX, 128, KXpad);
    cvt_wt_f16 <<<cdiv_l((long)128 * KXpad, 256), 256, 0, stream>>>(grw, wrT, KX, 128, KXpad);
    cvt_pad_f16<<<cdiv_l((long)B * FPKpad, 256), 256, 0, stream>>>(fpx, fpxp, B, FPK, FPKpad);
    cvt_wt_f16 <<<cdiv_l((long)64 * FPKpad, 256), 256, 0, stream>>>(fpw, fpwT, FPK, 64, FPKpad);

    // --- self-loop attrs = mean of incoming edge_attr per node
    loop_attr_acc<<<cdiv_l(E, 256), 256, 0, stream>>>(etgt, eattr, lsum, cnt, E);
    loop_attr_fin<<<cdiv_l((long)N * 12, 256), 256, 0, stream>>>(lsum, cnt, N);

    // --- node transforms via WMMA: xl = x@Wl+bl, xr = x@Wr+br  (N x 54 -> 128)
    {
        long waves  = ((long)N / 16) * (128 / 16);
        int  blocks = cdiv_l(waves * 32, 256);
        gemm_wmma_f16<<<blocks, 256, 0, stream>>>(xpad, wlT, glb, xl, N, KXpad, 128);
        gemm_wmma_f16<<<blocks, 256, 0, stream>>>(xpad, wrT, grb, xr, N, KXpad, 128);
    }

    // --- GATv2 attention: logits + softmax-max, exp + sum, weighted aggregation
    gat_edge_logit<<<cdiv_l((long)Etot * 32, 256), 256, 0, stream>>>(
        esrc, etgt, eattr, lsum, xl, xr, gew, gatt, elog, mx, E, N);
    gat_edge_exp<<<cdiv_l(Etot, 256), 256, 0, stream>>>(etgt, elog, mx, denom, E, N);
    gat_edge_aggr<<<cdiv_l((long)Etot * 32, 256), 256, 0, stream>>>(
        esrc, etgt, elog, denom, xl, outAcc, E, N);

    // --- node post (bias+BN+GELU) fused with global mean-pool scatter
    gat_node_post<<<cdiv_l((long)N * 128, 256), 256, 0, stream>>>(
        outAcc, gbias, bn4g, bn4b, bn4m, bn4v, batch, pool, gcnt, N);

    // --- fingerprint branch via WMMA: gelu(bn(fp_x@fp_w + fp_b))  (B x 3387 -> 64)
    {
        long waves  = ((long)B / 16) * (64 / 16);
        int  blocks = cdiv_l(waves * 32, 256);
        gemm_wmma_f16_bngelu<<<blocks, 256, 0, stream>>>(fpxp, fpwT, fpb,
                                                         fpbng, fpbnb, fpbnm, fpbnv,
                                                         fpact, B, FPKpad, 64);
    }

    // --- fusion head: z=[pool/cnt | fp | cluster] (202) -> 32 -> 1
    fuse_head<<<B, 32, 0, stream>>>(pool, gcnt, fpact, cx, f1w, f1b,
                                    fbng, fbnb, fbnm, fbnv, f2w, f2b,
                                    (float*)d_out);
}
